// AnswerDiscriminator_1494648619640
// MI455X (gfx1250) — compile-verified, matrix operand order checked by reading, not run on hardware
//
#include <hip/hip_runtime.h>
#include <hip/hip_bf16.h>
#include <math.h>
#include <stdint.h>

#define BB 32
#define TT 512
#define DD 768
#define HH 1024
#define G4 4096   // 4*H

typedef __hip_bfloat16 bf16;
typedef __attribute__((ext_vector_type(16))) __bf16 v16bf;
typedef __attribute__((ext_vector_type(8)))  float  v8f;

union FragU { v16bf v; uint4 q[2]; };

// ---- CDNA5 data-mover capability probes (compile-time) ---------------------
#if __has_builtin(__builtin_amdgcn_tensor_load_to_lds) && __has_builtin(__builtin_amdgcn_s_wait_tensorcnt)
#define USE_TDM 1
#else
#define USE_TDM 0
#endif
#if !USE_TDM && __has_builtin(__builtin_amdgcn_global_load_async_to_lds_b128) && __has_builtin(__builtin_amdgcn_s_wait_asynccnt)
#define USE_ASYNC 1
#else
#define USE_ASYNC 0
#endif

#if USE_TDM
typedef __attribute__((ext_vector_type(4))) unsigned int u32x4;
typedef __attribute__((ext_vector_type(8))) int i32x8;
typedef __attribute__((ext_vector_type(4))) int i32x4;

// Issue one TDM 2-D tile load (rows x cols bf16 elements, row stride in
// elements) from global into LDS at byte offset ldsOff. D# layout per
// CDNA5 ISA: group0 {count=1, lds_addr, global_addr,
// type=2}; group1 {data_size=2B, tensor dims, tile dims, dim0 stride}.
// This toolchain (clang-23 / therock-10.0) uses the 6-arg builtin:
//   (uint32x4 g0, int32x8 g1, int32x4 g2, int32x4 g3, int32x8, i32 cpol)
__device__ __forceinline__ void tdm_load_tile_2d(unsigned ldsOff, const void* g,
                                                 unsigned rows, unsigned cols,
                                                 unsigned strideElems) {
  unsigned long long ga = (unsigned long long)(uintptr_t)g;
  u32x4 g0;
  g0[0] = 1u;                                              // count=1, user D#
  g0[1] = ldsOff;                                          // lds_addr (bytes)
  g0[2] = (unsigned)ga;                                    // global_addr lo
  g0[3] = (unsigned)((ga >> 32) & 0x01FFFFFFull) | 0x80000000u;  // hi | type=2
  i32x8 g1;
  g1[0] = 1 << 16;                                         // data_size=1 (2B)
  g1[1] = (int)((cols & 0xFFFFu) << 16);                   // tensor_dim0 lo16
  g1[2] = (int)(((cols >> 16) & 0xFFFFu) | ((rows & 0xFFFFu) << 16));
  g1[3] = (int)(((rows >> 16) & 0xFFFFu) | ((cols & 0xFFFFu) << 16)); // tile_dim0
  g1[4] = (int)(rows & 0xFFFFu);                           // tile_dim1, tile_dim2=0
  g1[5] = (int)strideElems;                                // dim0 stride lo32
  g1[6] = 0;
  g1[7] = 0;
  i32x4 z4 = {0, 0, 0, 0};
  i32x8 z8 = {0, 0, 0, 0, 0, 0, 0, 0};
  __builtin_amdgcn_tensor_load_to_lds(g0, g1, z4, z4, z8, 0);
}
#endif

__device__ __forceinline__ unsigned lds_off_of(const void* p) {
  // generic LDS pointer: addr[31:0] is the LDS byte offset (aperture in hi bits)
  return (unsigned)(uintptr_t)p;
}

// Load a 16x32 bf16 WMMA operand fragment from row-major [row][K] global data.
// Lanes 0-15 -> K 0..7 & 16..23, lanes 16-31 -> K 8..15 & 24..31.
__device__ __forceinline__ v16bf load_frag16x32(const bf16* __restrict__ base,
                                                int row, int ldk, int k0, int lane) {
  const int klo = (lane & 16) ? 8 : 0;
  const bf16* p = base + (size_t)row * ldk + (k0 + klo);
  FragU u;
  u.q[0] = *reinterpret_cast<const uint4*>(p);
  u.q[1] = *reinterpret_cast<const uint4*>(p + 16);
  return u.v;
}

// Read the same fragment shape out of an LDS-resident 32x32 tile.
__device__ __forceinline__ v16bf lds_frag16x32(const bf16* tile, int row, int lane) {
  const int klo = (lane & 16) ? 8 : 0;
  FragU u;
  u.q[0] = *reinterpret_cast<const uint4*>(tile + row * 32 + klo);
  u.q[1] = *reinterpret_cast<const uint4*>(tile + row * 32 + 16 + klo);
  return u.v;
}

// ---------------- big input-projection GEMM: G[M][N] = A[M][K] * W[N][K]^T ---
// A 32x32 k-chunk of A is staged in double-buffered LDS (shared by all 4
// waves); W fragments are register double-buffered from global.
__global__ __launch_bounds__(128) void k_gemm_bf16(
    const bf16* __restrict__ A, const bf16* __restrict__ W,
    float* __restrict__ Gout, int K, int N) {
  __shared__ __align__(16) bf16 atile[2][32][32];
  const int lane = threadIdx.x & 31;
  const int wave = threadIdx.x >> 5;
  const int m0 = blockIdx.x * 32;
  const int n0 = blockIdx.y * 64 + wave * 16;
  const int rB  = n0 + (lane & 15);
  const int row = threadIdx.x >> 2;          // staging: 32 rows x 64B
  const int seg = (threadIdx.x & 3) * 8;     // 4 x 16B per row
  const int ra0 = lane & 15;
  const int ra1 = 16 + (lane & 15);

  auto stage = [&](int buf, int k0) {
#if USE_TDM
    if (wave == 0)
      tdm_load_tile_2d(lds_off_of(&atile[buf][0][0]),
                       A + (size_t)m0 * K + k0, 32u, 32u, (unsigned)K);
#elif USE_ASYNC
    {
      const bf16* src = A + (size_t)(m0 + row) * K + k0 + seg;
      __builtin_amdgcn_global_load_async_to_lds_b128(
          (__attribute__((address_space(1))) void*)(uintptr_t)src,
          (__attribute__((address_space(3))) void*)(uintptr_t)&atile[buf][row][seg],
          0, 0);
    }
#else
    *reinterpret_cast<uint4*>(&atile[buf][row][seg]) =
        *reinterpret_cast<const uint4*>(A + (size_t)(m0 + row) * K + k0 + seg);
#endif
  };
  auto wait_stage = [&]() {
#if USE_TDM
    if (wave == 0) __builtin_amdgcn_s_wait_tensorcnt((short)0);
#elif USE_ASYNC
    __builtin_amdgcn_s_wait_asynccnt((short)0);
#endif
  };

  v8f acc0 = {}; v8f acc1 = {};
  const int KT = K / 32;
  stage(0, 0);
  v16bf bw = load_frag16x32(W, rB, K, 0, lane);
  for (int kt = 0; kt < KT; ++kt) {
    wait_stage();
    __syncthreads();                       // stage(kt) visible; kt-1 reads done
    if (kt + 1 < KT) stage((kt + 1) & 1, (kt + 1) * 32);  // overlap with math
    if (kt + 2 < KT)
      __builtin_prefetch(W + (size_t)rB * K + (kt + 2) * 32, 0, 1);
    v16bf a0 = lds_frag16x32(&atile[kt & 1][0][0], ra0, lane);
    v16bf a1 = lds_frag16x32(&atile[kt & 1][0][0], ra1, lane);
    v16bf bcur = bw;
    if (kt + 1 < KT) bw = load_frag16x32(W, rB, K, (kt + 1) * 32, lane);
    acc0 = __builtin_amdgcn_wmma_f32_16x16x32_bf16(false, a0, false, bcur,
                                                   (short)0, acc0, false, false);
    acc1 = __builtin_amdgcn_wmma_f32_16x16x32_bf16(false, a1, false, bcur,
                                                   (short)0, acc1, false, false);
  }
  const int mo = (lane & 16) ? 8 : 0;
  const int nj = n0 + (lane & 15);
#pragma unroll
  for (int r = 0; r < 8; ++r) {
    Gout[(size_t)(m0 + r + mo) * N + nj]      = acc0[r];
    Gout[(size_t)(m0 + 16 + r + mo) * N + nj] = acc1[r];
  }
}

// ---------------- fused recurrent step: gates = h@Whh^T + Gpre + b; update ---
__global__ __launch_bounds__(128) void k_lstm_step(
    const bf16* __restrict__ hin,        // [B][H] bf16 (h_{t-1})
    const bf16* __restrict__ whh,        // [4H][H] bf16
    const float* __restrict__ Gpre,      // [B*T][4H] precomputed x@Wih^T
    const float* __restrict__ bih, const float* __restrict__ bhh,
    const int* __restrict__ len, int t,
    float* __restrict__ c,               // [B][H] f32 cell state
    bf16* __restrict__ hout,             // [B][H] bf16 (h_t)
    bf16* __restrict__ outp, int outStride, int outOff) {
  __shared__ float gq[4][BB][16];        // 4 gate quarters x batch x j-slice
  __shared__ __align__(16) bf16 atile[2][32][32];
  const int lane = threadIdx.x & 31;
  const int q = threadIdx.x >> 5;        // gate quarter (i,f,g,o)
  const int n0 = q * HH + blockIdx.x * 16;
  const int rB = n0 + (lane & 15);
  const int row = threadIdx.x >> 2;
  const int seg = (threadIdx.x & 3) * 8;
  const int ra0 = lane & 15;
  const int ra1 = 16 + (lane & 15);

  auto stage = [&](int buf, int k0) {
#if USE_TDM
    if (q == 0)
      tdm_load_tile_2d(lds_off_of(&atile[buf][0][0]), hin + k0, 32u, 32u, HH);
#elif USE_ASYNC
    {
      const bf16* src = hin + (size_t)row * HH + k0 + seg;
      __builtin_amdgcn_global_load_async_to_lds_b128(
          (__attribute__((address_space(1))) void*)(uintptr_t)src,
          (__attribute__((address_space(3))) void*)(uintptr_t)&atile[buf][row][seg],
          0, 0);
    }
#else
    *reinterpret_cast<uint4*>(&atile[buf][row][seg]) =
        *reinterpret_cast<const uint4*>(hin + (size_t)row * HH + k0 + seg);
#endif
  };
  auto wait_stage = [&]() {
#if USE_TDM
    if (q == 0) __builtin_amdgcn_s_wait_tensorcnt((short)0);
#elif USE_ASYNC
    __builtin_amdgcn_s_wait_asynccnt((short)0);
#endif
  };

  v8f acc0 = {}; v8f acc1 = {};
  const int KT = HH / 32;
  stage(0, 0);
  v16bf bw = load_frag16x32(whh, rB, HH, 0, lane);
  for (int kt = 0; kt < KT; ++kt) {
    wait_stage();
    __syncthreads();
    if (kt + 1 < KT) stage((kt + 1) & 1, (kt + 1) * 32);
    v16bf a0 = lds_frag16x32(&atile[kt & 1][0][0], ra0, lane);
    v16bf a1 = lds_frag16x32(&atile[kt & 1][0][0], ra1, lane);
    v16bf bcur = bw;
    if (kt + 1 < KT) bw = load_frag16x32(whh, rB, HH, (kt + 1) * 32, lane);
    acc0 = __builtin_amdgcn_wmma_f32_16x16x32_bf16(false, a0, false, bcur,
                                                   (short)0, acc0, false, false);
    acc1 = __builtin_amdgcn_wmma_f32_16x16x32_bf16(false, a1, false, bcur,
                                                   (short)0, acc1, false, false);
  }
  const int mo = (lane & 16) ? 8 : 0;
  const int j = lane & 15;
  const float bias = bih[rB] + bhh[rB];
#pragma unroll
  for (int r = 0; r < 8; ++r) {
    int m = r + mo;
    gq[q][m][j]  = acc0[r] + Gpre[(size_t)(m * TT + t) * G4 + rB] + bias;
    int m1 = 16 + r + mo;
    gq[q][m1][j] = acc1[r] + Gpre[(size_t)(m1 * TT + t) * G4 + rB] + bias;
  }
  __syncthreads();
  for (int e = threadIdx.x; e < BB * 16; e += 128) {
    int b = e >> 4, jj = e & 15;
    int jg = blockIdx.x * 16 + jj;
    float gi = gq[0][b][jj], gf = gq[1][b][jj];
    float gg = gq[2][b][jj], go = gq[3][b][jj];
    float iv = 1.f / (1.f + __expf(-gi));
    float fv = 1.f / (1.f + __expf(-gf));
    float gv = tanhf(gg);
    float ov = 1.f / (1.f + __expf(-go));
    float cold = c[b * HH + jg];
    float cn = fv * cold + iv * gv;
    float hn = ov * tanhf(cn);
    bool valid = t < len[b];
    float hold = __bfloat162float(hin[b * HH + jg]);
    float hv = valid ? hn : hold;
    c[b * HH + jg] = valid ? cn : cold;
    bf16 hb = __float2bfloat16(hv);
    hout[b * HH + jg] = hb;
    outp[(size_t)(b * TT + t) * outStride + outOff + jg] = hb;
  }
}

// ---------------- helpers ---------------------------------------------------
__global__ void k_len(const int* __restrict__ ids, int* __restrict__ len) {
  int b = threadIdx.x;
  if (b >= BB) return;
  int cnt = 0;
  for (int t = 0; t < TT; ++t) cnt += (ids[b * TT + t] != 0);
  len[b] = cnt;
}

__global__ void k_cvt(const float* __restrict__ s, bf16* __restrict__ d, size_t n) {
  size_t i = (size_t)blockIdx.x * blockDim.x + threadIdx.x;
  if (i < n) d[i] = __float2bfloat16(s[i]);
}

__global__ void k_zero32(unsigned int* __restrict__ p, size_t nw) {
  size_t i = (size_t)blockIdx.x * blockDim.x + threadIdx.x;
  if (i < nw) p[i] = 0u;
}

__global__ void k_embed(const int* __restrict__ ids, const float* __restrict__ am,
                        const float* __restrict__ emb, bf16* __restrict__ x) {
  size_t i = (size_t)blockIdx.x * blockDim.x + threadIdx.x;
  size_t total = (size_t)BB * TT * DD;
  if (i >= total) return;
  int k = (int)(i % DD);
  size_t bt = i / DD;
  int id = ids[bt];
  float e = emb[(size_t)id * DD + k];
  float m = am[bt];
  x[bt * (2 * DD) + k]      = __float2bfloat16(e);
  x[bt * (2 * DD) + DD + k] = __float2bfloat16(e * m);
}

__global__ void k_rev(const bf16* __restrict__ src, bf16* __restrict__ dst,
                      const int* __restrict__ len, int Din,
                      int srcStride, int dstStride, int dstOff) {
  size_t i = (size_t)blockIdx.x * blockDim.x + threadIdx.x;
  size_t total = (size_t)BB * TT * Din;
  if (i >= total) return;
  int k = (int)(i % Din);
  size_t bt = i / Din;
  int t = (int)(bt % TT);
  int b = (int)(bt / TT);
  int l = len[b];
  int ts = (t < l) ? (l - 1 - t) : t;
  dst[((size_t)b * TT + t) * dstStride + dstOff + k] =
      src[((size_t)b * TT + ts) * srcStride + k];
}

__global__ void k_feat(const bf16* __restrict__ hf, const bf16* __restrict__ hb,
                       float* __restrict__ feat) {
  int i = blockIdx.x * blockDim.x + threadIdx.x;
  if (i >= BB * 2 * HH) return;
  int b = i / (2 * HH);
  int k = i % (2 * HH);
  feat[i] = (k < HH) ? __bfloat162float(hf[b * HH + k])
                     : __bfloat162float(hb[b * HH + (k - HH)]);
}

__global__ void k_linbn(const float* __restrict__ X, const float* __restrict__ W,
                        const float* __restrict__ bias, const float* __restrict__ g,
                        const float* __restrict__ bb, float* __restrict__ Y, int K) {
  int j = blockIdx.x * blockDim.x + threadIdx.x;
  if (j >= HH) return;
  float v[BB];
  for (int b = 0; b < BB; ++b) {
    float s = bias[j];
    const float* xr = X + (size_t)b * K;
    const float* wr = W + (size_t)j * K;
    for (int k = 0; k < K; ++k) s += xr[k] * wr[k];
    v[b] = s;
  }
  float mu = 0.f;
  for (int b = 0; b < BB; ++b) mu += v[b];
  mu *= (1.f / BB);
  float var = 0.f;
  for (int b = 0; b < BB; ++b) { float d = v[b] - mu; var += d * d; }
  var *= (1.f / BB);
  float rs = rsqrtf(var + 1e-5f);
  for (int b = 0; b < BB; ++b) {
    float y = (v[b] - mu) * rs * g[j] + bb[j];
    Y[b * HH + j] = (y > 0.f) ? y : 0.2f * y;
  }
}

__global__ void k_final(const float* __restrict__ Y, const float* __restrict__ w,
                        const float* __restrict__ b, float* __restrict__ out) {
  int bi = threadIdx.x;
  if (bi >= BB) return;
  float s = b[0];
  for (int k = 0; k < HH; ++k) s += Y[bi * HH + k] * w[k];
  out[bi] = s;
}

// ---------------- host driver ----------------------------------------------
extern "C" void kernel_launch(void* const* d_in, const int* in_sizes, int n_in,
                              void* d_out, int out_size, void* d_ws, size_t ws_size,
                              hipStream_t stream) {
  (void)in_sizes; (void)n_in; (void)out_size; (void)ws_size;
  const int*   c_ids     = (const int*)d_in[0];
  const float* a_mask    = (const float*)d_in[1];
  const float* embedding = (const float*)d_in[2];
  const float *wih[2][2], *whh[2][2], *bih[2][2], *bhh[2][2];
  int idx = 3;
  for (int l = 0; l < 2; ++l)
    for (int d = 0; d < 2; ++d) {
      wih[l][d] = (const float*)d_in[idx++];
      whh[l][d] = (const float*)d_in[idx++];
      bih[l][d] = (const float*)d_in[idx++];
      bhh[l][d] = (const float*)d_in[idx++];
    }
  const float* lin1_w = (const float*)d_in[idx++];
  const float* lin1_b = (const float*)d_in[idx++];
  const float* bn1_g  = (const float*)d_in[idx++];
  const float* bn1_b  = (const float*)d_in[idx++];
  const float* lin2_w = (const float*)d_in[idx++];
  const float* lin2_b = (const float*)d_in[idx++];
  const float* bn2_g  = (const float*)d_in[idx++];
  const float* bn2_b  = (const float*)d_in[idx++];
  const float* lin3_w = (const float*)d_in[idx++];
  const float* lin3_b = (const float*)d_in[idx++];
  float* out = (float*)d_out;

  char* base = (char*)d_ws;
  size_t off = 0;
  auto alloc = [&](size_t bytes) -> void* {
    off = (off + 255) & ~(size_t)255;
    void* p = base + off;
    off += bytes;
    return p;
  };

  const int din[2] = {2 * DD, 2 * HH};
  bf16 *wih_bf[2][2], *whh_bf[2][2];
  for (int l = 0; l < 2; ++l)
    for (int d = 0; d < 2; ++d) {
      wih_bf[l][d] = (bf16*)alloc((size_t)G4 * din[l] * 2);
      whh_bf[l][d] = (bf16*)alloc((size_t)G4 * HH * 2);
    }
  bf16* x0  = (bf16*)alloc((size_t)BB * TT * (2 * DD) * 2);
  bf16* x0r = (bf16*)alloc((size_t)BB * TT * (2 * DD) * 2);
  bf16* x1  = (bf16*)alloc((size_t)BB * TT * (2 * HH) * 2);
  bf16* x1r = (bf16*)alloc((size_t)BB * TT * (2 * HH) * 2);
  float* Gf = (float*)alloc((size_t)BB * TT * G4 * 4);
  float* Gb = (float*)alloc((size_t)BB * TT * G4 * 4);
  bf16* obr = (bf16*)alloc((size_t)BB * TT * HH * 2);
  bf16* hpp[2][2];
  float* cst[2];
  for (int d = 0; d < 2; ++d) {
    hpp[d][0] = (bf16*)alloc((size_t)BB * HH * 2);
    hpp[d][1] = (bf16*)alloc((size_t)BB * HH * 2);
    cst[d]    = (float*)alloc((size_t)BB * HH * 4);
  }
  int*   len  = (int*)alloc(BB * 4);
  float* feat = (float*)alloc((size_t)BB * 2 * HH * 4);
  float* y1   = (float*)alloc((size_t)BB * HH * 4);
  float* y2   = (float*)alloc((size_t)BB * HH * 4);

  k_len<<<1, 32, 0, stream>>>(c_ids, len);

  for (int l = 0; l < 2; ++l)
    for (int d = 0; d < 2; ++d) {
      size_t n1 = (size_t)G4 * din[l];
      k_cvt<<<(unsigned)((n1 + 255) / 256), 256, 0, stream>>>(wih[l][d], wih_bf[l][d], n1);
      size_t n2 = (size_t)G4 * HH;
      k_cvt<<<(unsigned)((n2 + 255) / 256), 256, 0, stream>>>(whh[l][d], whh_bf[l][d], n2);
    }

  {
    size_t tot = (size_t)BB * TT * DD;
    k_embed<<<(unsigned)((tot + 255) / 256), 256, 0, stream>>>(c_ids, a_mask, embedding, x0);
  }

  auto zero_states = [&](int d) {
    size_t hw = (size_t)BB * HH / 2;
    k_zero32<<<(unsigned)((hw + 255) / 256), 256, 0, stream>>>((unsigned int*)hpp[d][0], hw);
    k_zero32<<<(unsigned)((hw + 255) / 256), 256, 0, stream>>>((unsigned int*)hpp[d][1], hw);
    size_t cw = (size_t)BB * HH;
    k_zero32<<<(unsigned)((cw + 255) / 256), 256, 0, stream>>>((unsigned int*)cst[d], cw);
  };

  auto run_dir = [&](int l, int d, const bf16* xin, int K, float* Gbuf,
                     bf16* outp, int outStride, int outOff) {
    dim3 gg(BB * TT / 32, G4 / 64);
    k_gemm_bf16<<<gg, 128, 0, stream>>>(xin, wih_bf[l][d], Gbuf, K, G4);
    zero_states(d);
    for (int t = 0; t < TT; ++t) {
      const bf16* hi = (t & 1) ? hpp[d][1] : hpp[d][0];
      bf16*       ho = (t & 1) ? hpp[d][0] : hpp[d][1];
      k_lstm_step<<<HH / 16, 128, 0, stream>>>(hi, whh_bf[l][d], Gbuf,
                                               bih[l][d], bhh[l][d], len, t,
                                               cst[d], ho, outp, outStride, outOff);
    }
  };

  size_t revtot;

  // ---- layer 0 ----
  revtot = (size_t)BB * TT * (2 * DD);
  k_rev<<<(unsigned)((revtot + 255) / 256), 256, 0, stream>>>(x0, x0r, len,
                                                              2 * DD, 2 * DD, 2 * DD, 0);
  run_dir(0, 0, x0,  2 * DD, Gf, x1, 2 * HH, 0);
  run_dir(0, 1, x0r, 2 * DD, Gb, obr, HH, 0);
  revtot = (size_t)BB * TT * HH;
  k_rev<<<(unsigned)((revtot + 255) / 256), 256, 0, stream>>>(obr, x1, len,
                                                              HH, HH, 2 * HH, HH);

  // ---- layer 1 ---- (only final states feed the head)
  revtot = (size_t)BB * TT * (2 * HH);
  k_rev<<<(unsigned)((revtot + 255) / 256), 256, 0, stream>>>(x1, x1r, len,
                                                              2 * HH, 2 * HH, 2 * HH, 0);
  run_dir(1, 0, x1,  2 * HH, Gf, obr, HH, 0);
  run_dir(1, 1, x1r, 2 * HH, Gb, obr, HH, 0);

  // ---- MLP head on final hidden states (slot 0 after even T) ----
  k_feat<<<(BB * 2 * HH + 255) / 256, 256, 0, stream>>>(hpp[0][0], hpp[1][0], feat);
  k_linbn<<<4, 256, 0, stream>>>(feat, lin1_w, lin1_b, bn1_g, bn1_b, y1, 2 * HH);
  k_linbn<<<4, 256, 0, stream>>>(y1, lin2_w, lin2_b, bn2_g, bn2_b, y2, HH);
  k_final<<<1, 32, 0, stream>>>(y2, lin3_w, lin3_b, out);
}
